// update_eT_13469017440641
// MI455X (gfx1250) — matrix operand order repaired and checked by reading, not compiled, not run on hardware
//
#include <hip/hip_runtime.h>
#include <hip/hip_bf16.h>

typedef _Float16 v16h __attribute__((ext_vector_type(16)));
typedef _Float16 v8h  __attribute__((ext_vector_type(8)));
typedef float    v8f  __attribute__((ext_vector_type(8)));
typedef unsigned int u32x4 __attribute__((ext_vector_type(4)));
typedef int      i32x4 __attribute__((ext_vector_type(4)));
typedef int      i32x8 __attribute__((ext_vector_type(8)));

#if defined(__has_builtin)
#if __has_builtin(__builtin_amdgcn_tensor_load_to_lds) && __has_builtin(__builtin_amdgcn_s_wait_tensorcnt)
#define USE_TDM 1
#endif
#endif

// silu via v_exp + v_rcp (avoid IEEE-div expansion)
__device__ __forceinline__ float silu_f(float x) {
  return x * __builtin_amdgcn_rcpf(1.0f + __expf(-x));
}

__device__ __forceinline__ v8f wmma16(v16h a, v16h b, v8f c) {
  return __builtin_amdgcn_wmma_f32_16x16x32_f16(false, a, false, b, (short)0, c, false, false);
}

// A fragment (16x32, f16) from row-major f32 src: lane m = lane&15, half = lane>>4;
// element e -> k = k0 + (e>>3)*16 + half*8 + (e&7)  (two contiguous 32B runs).
__device__ __forceinline__ v16h load_afrag_f32(const float* __restrict__ src, int ld, int row0, int k0) {
  int lane = threadIdx.x & 31, m = lane & 15, half = lane >> 4;
  const float* p = src + (size_t)(row0 + m) * ld + k0 + half * 8;
  v8f c0 = *(const v8f*)p;
  v8f c1 = *(const v8f*)(p + 16);
  v16h r;
#pragma unroll
  for (int i = 0; i < 8; ++i) { r[i] = (_Float16)c0[i]; r[8 + i] = (_Float16)c1[i]; }
  return r;
}

// Same but from row-major f16 source (global or LDS).
__device__ __forceinline__ v16h load_afrag_f16(const _Float16* __restrict__ src, int ld, int row0, int k0) {
  int lane = threadIdx.x & 31, m = lane & 15, half = lane >> 4;
  const _Float16* p = src + (size_t)(row0 + m) * ld + k0 + half * 8;
  v8h c0 = *(const v8h*)p;
  v8h c1 = *(const v8h*)(p + 16);
  v16h r;
#pragma unroll
  for (int i = 0; i < 8; ++i) { r[i] = c0[i]; r[8 + i] = c1[i]; }
  return r;
}

// B fragment (32x16) from K-major f16 weights Bt[n][k]: lane n = lane&15;
// element e -> k = k0 + (e>>3)*16 + (lane>>4)*8 + (e&7) (two 16B runs).
__device__ __forceinline__ v16h load_bfrag(const _Float16* __restrict__ Bt, int ldk, int n0, int k0) {
  int lane = threadIdx.x & 31;
  const _Float16* p = Bt + (size_t)(n0 + (lane & 15)) * ldk + k0 + (lane >> 4) * 8;
  v8h lo = *(const v8h*)p;
  v8h hi = *(const v8h*)(p + 16);
  v16h r;
#pragma unroll
  for (int i = 0; i < 8; ++i) { r[i] = lo[i]; r[8 + i] = hi[i]; }
  return r;
}

// ---------------------------------------------------------------------------
// Kernel 0: weight prep (f32->f16 transposes), fused rbf table, agg zeroing.
// ---------------------------------------------------------------------------
__global__ void prep_kernel(const float* __restrict__ rbf0, const float* __restrict__ W_ji,
                            const float* __restrict__ W_kj, const float* __restrict__ W_rbf1,
                            const float* __restrict__ W_rbf2, const float* __restrict__ W_sbf1,
                            const float* __restrict__ W_down, const float* __restrict__ W_up,
                            const float* __restrict__ W_bil,
                            _Float16* Wjit, _Float16* Wkjt, _Float16* Wdnt, _Float16* Wupt,
                            _Float16* Wst, _Float16* Wbil16, float* rbfw, float* agg, int E) {
  int tid = blockIdx.x * blockDim.x + threadIdx.x;
  int nth = gridDim.x * blockDim.x;
  for (int i = tid; i < 128 * 128; i += nth) {
    int n = i >> 7, k = i & 127;
    Wjit[n * 128 + k] = (_Float16)W_ji[k * 128 + n];
    Wkjt[n * 128 + k] = (_Float16)W_kj[k * 128 + n];
  }
  for (int i = tid; i < 64 * 128; i += nth) {  // W_down [128,64] -> [n=64][k=128]
    int n = i >> 7, k = i & 127;
    Wdnt[n * 128 + k] = (_Float16)W_down[k * 64 + n];
  }
  for (int i = tid; i < 128 * 64; i += nth) {  // W_up [64,128] -> [n=128][k=64]
    int n = i >> 6, k = i & 63;
    Wupt[n * 64 + k] = (_Float16)W_up[k * 128 + n];
  }
  for (int i = tid; i < 64 * 64; i += nth) {   // W_sbf1 [42,64] -> [n=64][k=64] zero-padded
    int n = i >> 6, k = i & 63;
    Wst[n * 64 + k] = (k < 42) ? (_Float16)W_sbf1[k * 64 + n] : (_Float16)0.0f;
  }
  for (int i = tid; i < 64 * 64 * 64; i += nth)  // W_bil[i][j][l] flat == Bt[n=i][k=j*64+l]
    Wbil16[i] = (_Float16)W_bil[i];
  for (int i = tid; i < E * 128; i += nth) {   // rbf = rbf0 @ (W_rbf1 @ W_rbf2), fused
    int e = i >> 7, n = i & 127;
    float s = 0.0f;
#pragma unroll
    for (int q = 0; q < 6; ++q) {
      float wqn = 0.0f;
#pragma unroll
      for (int b = 0; b < 8; ++b) wqn += W_rbf1[q * 8 + b] * W_rbf2[b * 128 + n];
      s += rbf0[e * 6 + q] * wqn;
    }
    rbfw[i] = s;
  }
  for (int i = tid; i < E * 64; i += nth) agg[i] = 0.0f;
}

// ---------------------------------------------------------------------------
// Kernel 1: per 16-edge tile (1 wave): x_ji -> out, x_kj*rbf -> x_down (f16).
// ---------------------------------------------------------------------------
__global__ void __launch_bounds__(32)
edge_kernel(const float* __restrict__ x1, const float* __restrict__ b_ji,
            const float* __restrict__ b_kj, const _Float16* __restrict__ Wjit,
            const _Float16* __restrict__ Wkjt, const _Float16* __restrict__ Wdnt,
            const float* __restrict__ rbfw, float* __restrict__ out,
            _Float16* __restrict__ xdown) {
  __shared__ __align__(32) _Float16 tile[16 * 128];
  int e0 = blockIdx.x * 16;
  int lane = threadIdx.x & 31, nlo = lane & 15, half = lane >> 4;

  v16h a[4];
#pragma unroll
  for (int c = 0; c < 4; ++c) a[c] = load_afrag_f32(x1, 128, e0, c * 32);

  // x_ji = silu(x1 @ W_ji + b_ji) -> out
#pragma unroll
  for (int nt = 0; nt < 8; ++nt) {
    v8f acc = {};
#pragma unroll
    for (int c = 0; c < 4; ++c) acc = wmma16(a[c], load_bfrag(Wjit, 128, nt * 16, c * 32), acc);
    float bias = b_ji[nt * 16 + nlo];
#pragma unroll
    for (int r = 0; r < 8; ++r) {
      int m2 = half * 8 + r;
      out[(size_t)(e0 + m2) * 128 + nt * 16 + nlo] = silu_f(acc[r] + bias);
    }
  }

  // x_kj = silu(x1 @ W_kj + b_kj) * rbf -> LDS tile (f16)
#pragma unroll
  for (int nt = 0; nt < 8; ++nt) {
    v8f acc = {};
#pragma unroll
    for (int c = 0; c < 4; ++c) acc = wmma16(a[c], load_bfrag(Wkjt, 128, nt * 16, c * 32), acc);
    float bias = b_kj[nt * 16 + nlo];
#pragma unroll
    for (int r = 0; r < 8; ++r) {
      int m2 = half * 8 + r;
      float v = silu_f(acc[r] + bias) * rbfw[(size_t)(e0 + m2) * 128 + nt * 16 + nlo];
      tile[m2 * 128 + nt * 16 + nlo] = (_Float16)v;
    }
  }
  __syncthreads();

  // x_down = silu(x_kj @ W_down) -> global f16 [E,64]
  v16h a2[4];
#pragma unroll
  for (int c = 0; c < 4; ++c) a2[c] = load_afrag_f16(tile, 128, 0, c * 32);
#pragma unroll
  for (int nt = 0; nt < 4; ++nt) {
    v8f acc = {};
#pragma unroll
    for (int c = 0; c < 4; ++c) acc = wmma16(a2[c], load_bfrag(Wdnt, 128, nt * 16, c * 32), acc);
#pragma unroll
    for (int r = 0; r < 8; ++r) {
      int m2 = half * 8 + r;
      xdown[(size_t)(e0 + m2) * 64 + nt * 16 + nlo] = (_Float16)silu_f(acc[r]);
    }
  }
}

// ---------------------------------------------------------------------------
// Kernel 2: sbf_p = sbf @ W_sbf1 (K=42 padded to 64) -> f16 [T,64].
// ---------------------------------------------------------------------------
__global__ void __launch_bounds__(32)
sbfp_kernel(const float* __restrict__ sbf, const _Float16* __restrict__ Wst,
            _Float16* __restrict__ sbfp) {
  int t0 = blockIdx.x * 16;
  int lane = threadIdx.x & 31, m = lane & 15, nlo = lane & 15, half = lane >> 4;
  v16h a[2];
#pragma unroll
  for (int c = 0; c < 2; ++c) {
#pragma unroll
    for (int e = 0; e < 16; ++e) {
      int k = c * 32 + ((e >> 3) * 16) + half * 8 + (e & 7);
      a[c][e] = (k < 42) ? (_Float16)sbf[(size_t)(t0 + m) * 42 + k] : (_Float16)0.0f;
    }
  }
#pragma unroll
  for (int nt = 0; nt < 4; ++nt) {
    v8f acc = {};
#pragma unroll
    for (int c = 0; c < 2; ++c) acc = wmma16(a[c], load_bfrag(Wst, 64, nt * 16, c * 32), acc);
#pragma unroll
    for (int r = 0; r < 8; ++r)
      sbfp[(size_t)(t0 + half * 8 + r) * 64 + nt * 16 + nlo] = (_Float16)acc[r];
  }
}

// ---------------------------------------------------------------------------
// Kernel 3: bilinear einsum as GEMM [T,4096]x[4096,64] with on-the-fly A,
// K split over gridDim.y (j-halves), B half (256KB) staged in LDS via TDM.
// Each wave carries TWO 16-row M-tiles so every LDS B-fragment feeds 2 WMMAs.
// Partial results merged with f32 atomics (== segment_sum by idx_ji).
// ---------------------------------------------------------------------------
__global__ void __launch_bounds__(256)
einsum_kernel(const _Float16* __restrict__ Wbil16, const _Float16* __restrict__ sbfp,
              const _Float16* __restrict__ xdown, const int* __restrict__ idx_kj,
              const int* __restrict__ idx_ji, float* __restrict__ agg, int ntiles) {
  extern __shared__ _Float16 smem[];
  _Float16* Bl = smem;  // [64][2048] f16 = 256KB, rows n=0..63, k'=0..2047
  const int jh = blockIdx.y;           // j-half: j in [jh*32, jh*32+32)
  const int wave = threadIdx.x >> 5;
  const int lane = threadIdx.x & 31, m = lane & 15, half = lane >> 4;

#if defined(USE_TDM)
  if (threadIdx.x == 0) {
    unsigned long long ga = (unsigned long long)(const void*)(Wbil16 + (size_t)jh * 2048);
    unsigned int lds = (unsigned int)(unsigned long long)(void*)Bl;
    // D# group0: count=1, lds_addr, global_addr, type=2
    u32x4 g0 = {1u, lds, (unsigned int)ga,
                (unsigned int)((ga >> 32) & 0x01FFFFFFull) | 0x80000000u};
    // group1: data_size=8B; tensor_dim0=1024 (8B units/row), tensor_dim1=64,
    // tile_dim0=512 (=4KB row), tile_dim1=64, dim0_stride=1024 (=8KB)
    i32x8 g1 = {(int)(3u << 16), (int)(1024u << 16), (int)(64u << 16),
                (int)(512u << 16), 64, 1024, 0, 0};
    i32x4 gz = {0, 0, 0, 0};
#if defined(__clang_major__) && (__clang_major__ >= 23)
    i32x8 g4 = {0, 0, 0, 0, 0, 0, 0, 0};
    __builtin_amdgcn_tensor_load_to_lds(g0, g1, gz, gz, g4, 0);
#else
    __builtin_amdgcn_tensor_load_to_lds(g0, g1, gz, gz, 0);
#endif
    __builtin_amdgcn_s_wait_tensorcnt(0);
  }
#else
  for (int i = threadIdx.x; i < 64 * 2048 / 8; i += blockDim.x) {
    int n = i >> 8, c = i & 255;  // 256 v8h per row
    ((v8h*)Bl)[i] = *(const v8h*)(Wbil16 + (size_t)n * 4096 + (size_t)jh * 2048 + c * 8);
  }
#endif
  __syncthreads();

  const int npairs = (ntiles + 1) >> 1;
  for (int p = blockIdx.x * 8 + wave; p < npairs; p += gridDim.x * 8) {
    const int tA = p * 2;
    const int tB = tA + 1;
    const bool hasB = (tB < ntiles);
    const int t0a = tA * 16;
    const int t0b = hasB ? tB * 16 : t0a;

    const int ikjA = idx_kj[t0a + m];
    const int ikjB = idx_kj[t0b + m];

    // xg A-fragments for l in [0,32) and [32,64), reused across all 32 j's
    v16h xfA[2], xfB[2];
#pragma unroll
    for (int h = 0; h < 2; ++h) {
      const _Float16* pa = xdown + (size_t)ikjA * 64 + h * 32 + half * 8;
      const _Float16* pb = xdown + (size_t)ikjB * 64 + h * 32 + half * 8;
      v8h a0 = *(const v8h*)pa, a1 = *(const v8h*)(pa + 16);
      v8h b0 = *(const v8h*)pb, b1 = *(const v8h*)(pb + 16);
#pragma unroll
      for (int i = 0; i < 8; ++i) {
        xfA[h][i] = a0[i]; xfA[h][8 + i] = a1[i];
        xfB[h][i] = b0[i]; xfB[h][8 + i] = b1[i];
      }
    }

    // each lane's 32 sbf_p scalars (its row m, j-half jh) for both tiles
    v8h srA[4], srB[4];
    {
      const v8h* sa = (const v8h*)(sbfp + (size_t)(t0a + m) * 64 + jh * 32);
      const v8h* sb = (const v8h*)(sbfp + (size_t)(t0b + m) * 64 + jh * 32);
#pragma unroll
      for (int q = 0; q < 4; ++q) { srA[q] = sa[q]; srB[q] = sb[q]; }
    }

    v8f accA[4] = {}, accB[4] = {};
#pragma unroll
    for (int jp = 0; jp < 32; ++jp) {
      _Float16 sA = srA[jp >> 3][jp & 7];
      _Float16 sB = srB[jp >> 3][jp & 7];
#pragma unroll
      for (int h = 0; h < 2; ++h) {
        v16h afA, afB;
#pragma unroll
        for (int i = 0; i < 16; ++i) { afA[i] = sA * xfA[h][i]; afB[i] = sB * xfB[h][i]; }
        int k0 = (jp * 2 + h) * 32;
#pragma unroll
        for (int nt = 0; nt < 4; ++nt) {
          v16h bf = load_bfrag(Bl, 2048, nt * 16, k0);  // one LDS read, two WMMAs
          accA[nt] = wmma16(afA, bf, accA[nt]);
          accB[nt] = wmma16(afB, bf, accB[nt]);
        }
      }
    }

    // scatter-add partial x_t into agg[idx_ji] (this IS the segment_sum)
#pragma unroll
    for (int r = 0; r < 8; ++r) {
      int m2 = half * 8 + r;
      int ejA = idx_ji[t0a + m2];
#pragma unroll
      for (int nt = 0; nt < 4; ++nt)
        unsafeAtomicAdd(agg + (size_t)ejA * 64 + nt * 16 + (lane & 15), accA[nt][r]);
    }
    if (hasB) {
#pragma unroll
      for (int r = 0; r < 8; ++r) {
        int m2 = half * 8 + r;
        int ejB = idx_ji[t0b + m2];
#pragma unroll
        for (int nt = 0; nt < 4; ++nt)
          unsafeAtomicAdd(agg + (size_t)ejB * 64 + nt * 16 + (lane & 15), accB[nt][r]);
      }
    }
  }
}

// ---------------------------------------------------------------------------
// Kernel 4: out += silu(agg @ W_up)
// ---------------------------------------------------------------------------
__global__ void __launch_bounds__(32)
final_kernel(const float* __restrict__ agg, const _Float16* __restrict__ Wupt,
             float* __restrict__ out) {
  int e0 = blockIdx.x * 16;
  int lane = threadIdx.x & 31, nlo = lane & 15, half = lane >> 4;
  v16h a[2];
#pragma unroll
  for (int c = 0; c < 2; ++c) a[c] = load_afrag_f32(agg, 64, e0, c * 32);
#pragma unroll
  for (int nt = 0; nt < 8; ++nt) {
    v8f acc = {};
#pragma unroll
    for (int c = 0; c < 2; ++c) acc = wmma16(a[c], load_bfrag(Wupt, 64, nt * 16, c * 32), acc);
#pragma unroll
    for (int r = 0; r < 8; ++r) {
      int m2 = half * 8 + r;
      size_t off = (size_t)(e0 + m2) * 128 + nt * 16 + nlo;
      out[off] = out[off] + silu_f(acc[r]);
    }
  }
}

// ---------------------------------------------------------------------------
extern "C" void kernel_launch(void* const* d_in, const int* in_sizes, int n_in,
                              void* d_out, int out_size, void* d_ws, size_t ws_size,
                              hipStream_t stream) {
  const float* x1     = (const float*)d_in[0];
  const float* rbf0   = (const float*)d_in[1];
  const float* sbf    = (const float*)d_in[2];
  const int*   idx_kj = (const int*)d_in[3];
  const int*   idx_ji = (const int*)d_in[4];
  const float* W_ji   = (const float*)d_in[5];
  const float* b_ji   = (const float*)d_in[6];
  const float* W_kj   = (const float*)d_in[7];
  const float* b_kj   = (const float*)d_in[8];
  const float* W_rbf1 = (const float*)d_in[9];
  const float* W_rbf2 = (const float*)d_in[10];
  const float* W_sbf1 = (const float*)d_in[11];
  const float* W_down = (const float*)d_in[12];
  const float* W_up   = (const float*)d_in[13];
  const float* W_bil  = (const float*)d_in[14];
  float* out = (float*)d_out;

  const int E = in_sizes[0] / 128;
  const int T = in_sizes[2] / 42;

  char* ws = (char*)d_ws;
  size_t o = 0;
  _Float16* Wjit   = (_Float16*)(ws + o); o += 128 * 128 * 2;
  _Float16* Wkjt   = (_Float16*)(ws + o); o += 128 * 128 * 2;
  _Float16* Wdnt   = (_Float16*)(ws + o); o += 64 * 128 * 2;
  _Float16* Wupt   = (_Float16*)(ws + o); o += 128 * 64 * 2;
  _Float16* Wst    = (_Float16*)(ws + o); o += 64 * 64 * 2;
  _Float16* Wbil16 = (_Float16*)(ws + o); o += (size_t)64 * 64 * 64 * 2;
  float*    rbfw   = (float*)(ws + o);    o += (size_t)E * 128 * 4;
  _Float16* xdown  = (_Float16*)(ws + o); o += (size_t)E * 64 * 2;
  _Float16* sbfp   = (_Float16*)(ws + o); o += (size_t)T * 64 * 2;
  float*    agg    = (float*)(ws + o);    o += (size_t)E * 64 * 4;
  (void)ws_size; (void)n_in; (void)out_size;

  prep_kernel<<<2048, 256, 0, stream>>>(rbf0, W_ji, W_kj, W_rbf1, W_rbf2, W_sbf1,
                                        W_down, W_up, W_bil, Wjit, Wkjt, Wdnt, Wupt,
                                        Wst, Wbil16, rbfw, agg, E);
  edge_kernel<<<E / 16, 32, 0, stream>>>(x1, b_ji, b_kj, Wjit, Wkjt, Wdnt, rbfw, out, xdown);
  sbfp_kernel<<<T / 16, 32, 0, stream>>>(sbf, Wst, sbfp);

  const int lds_bytes = 64 * 2048 * 2;  // 256KB B-half
  (void)hipFuncSetAttribute((const void*)einsum_kernel,
                            hipFuncAttributeMaxDynamicSharedMemorySize, lds_bytes);
  einsum_kernel<<<dim3(192, 2, 1), 256, lds_bytes, stream>>>(Wbil16, sbfp, xdown,
                                                             idx_kj, idx_ji, agg, T / 16);
  final_kernel<<<E / 16, 32, 0, stream>>>(agg, Wupt, out);
}
